// DeepLatent_53188874993797
// MI455X (gfx1250) — matrix operand order repaired and checked by reading, not compiled, not run on hardware
//
#include <hip/hip_runtime.h>

typedef __attribute__((ext_vector_type(16))) __bf16 v16bf;
typedef __attribute__((ext_vector_type(8)))  __bf16 v8bf;
typedef __attribute__((ext_vector_type(8)))  float  v8f;

#define TILE_M  32
#define NWAVES  8
#define THREADS 256
#define K1PAD   288
#define LATENT  256
#define NPTS    (32 * 1024)
#define BATCHES 32
#define NSAMP   1024

__device__ __forceinline__ v16bf cat8(v8bf lo, v8bf hi) {
  return __builtin_shufflevector(lo, hi, 0,1,2,3,4,5,6,7,8,9,10,11,12,13,14,15);
}

// dst[TILE_M, N] = act(src[TILE_M, K] @ W + bias), W given pre-transposed as
// Wt[N, K] bf16 row-major so B-matrix lanes read 32 contiguous bytes.
// A operand k order used by this code (16x32): lane L holds row M=L%16,
// K = 16*(L/16)..+15 contiguous. B operand, same k order as A (32x16): lane L holds col N=L%16,
// K = 16*(L/16)..+15 contiguous (2 bf16 per VGPR).
template<int K, int N, bool RELU, bool F32OUT>
__device__ __forceinline__ void layer_wmma(const __bf16* src, void* dstv,
                                           const __bf16* __restrict__ Wt,
                                           const float*  __restrict__ bias,
                                           int wave, int lane) {
  constexpr int MT = TILE_M / 16;
  constexpr int NT = N / 16;
  constexpr int KS = K / 32;
  const int lmod = lane & 15, ldiv = lane >> 4;
  for (int t = wave; t < MT * NT; t += NWAVES) {   // uniform trip count per wave
    const int mt = t % MT, nt = t / MT;
    const __bf16* arow = src + (mt * 16 + lmod) * K + 16 * ldiv;
    const __bf16* brow = Wt + (size_t)(nt * 16 + lmod) * K + 16 * ldiv;
    v8f acc = {};
#pragma unroll
    for (int kc = 0; kc < KS; ++kc) {
      v8bf a0 = *(const v8bf*)(arow + kc * 32);        // ds_load_b128
      v8bf a1 = *(const v8bf*)(arow + kc * 32 + 8);    // ds_load_b128
      v16bf av = cat8(a0, a1);
      v16bf bv = *(const v16bf*)(brow + kc * 32);      // global 32B (L2-resident)
      acc = __builtin_amdgcn_wmma_f32_16x16x32_bf16(
          /*neg_a=*/false, av, /*neg_b=*/false, bv,
          /*c_mod=*/(short)0, acc, /*reuse_a=*/false, /*reuse_b=*/false);
    }
    const float bsc  = bias[nt * 16 + lmod];
    const int mbase  = mt * 16 + 8 * ldiv;
    const int ncol   = nt * 16 + lmod;
#pragma unroll
    for (int r = 0; r < 8; ++r) {          // C/D layout: VGPR r -> M = r + 8*(L/16)
      float v = acc[r] + bsc;
      if (RELU) v = fmaxf(v, 0.0f);
      if (F32OUT) ((float*)dstv)[(mbase + r) * N + ncol] = v;
      else        ((__bf16*)dstv)[(mbase + r) * N + ncol] = (__bf16)v;
    }
  }
}

// ---- weight transpose + bf16 convert (runs every launch; tiny) ----
__global__ void prep_kernel(const float* __restrict__ W1, const float* __restrict__ W2,
                            const float* __restrict__ W3,
                            __bf16* __restrict__ W1t, __bf16* __restrict__ W2t,
                            __bf16* __restrict__ W3t) {
  const int N1 = 512 * K1PAD, N2 = 512 * 512, N3 = 256 * 512;
  int i = blockIdx.x * THREADS + threadIdx.x;
  if (i < N1) {                      // W1t[n][k], k padded 259->288 with zeros
    int n = i / K1PAD, k = i - n * K1PAD;
    W1t[i] = (k < 259) ? (__bf16)W1[k * 512 + n] : (__bf16)0.0f;
  } else if (i < N1 + N2) {          // W2t[n][k] = W2[k][n]
    int j = i - N1, n = j >> 9, k = j & 511;
    W2t[j] = (__bf16)W2[k * 512 + n];
  } else if (i < N1 + N2 + N3) {     // W3t[n][k] = W3[k][n]  (W3: [512,256])
    int j = i - N1 - N2, n = j >> 9, k = j & 511;
    W3t[j] = (__bf16)W3[k * 256 + n];
  }
}

// ---- fused 4-layer MLP, one block per 32 points ----
__global__ __launch_bounds__(THREADS)
void mlp_kernel(const float* __restrict__ obs, const float* __restrict__ latent,
                const __bf16* __restrict__ W1t, const float* __restrict__ b1,
                const __bf16* __restrict__ W2t, const float* __restrict__ b2,
                const __bf16* __restrict__ W3t, const float* __restrict__ b3,
                const float* __restrict__ W4,  const float* __restrict__ b4,
                float* __restrict__ obs_est) {
  __shared__ __align__(32) __bf16 s_x [TILE_M * K1PAD];   // 18 KB
  __shared__ __align__(32) __bf16 s_h1[TILE_M * 512];     // 32 KB
  __shared__ __align__(32) __bf16 s_h2[TILE_M * 512];     // 32 KB
  float* s_h3 = (float*)s_h1;   // 32 KB f32 alias; h1 dead after layer 2

  const int tid = threadIdx.x, wave = tid >> 5, lane = tid & 31;
  const int p0 = blockIdx.x * TILE_M;
  const int batch = p0 >> 10;   // 1024 points per batch; TILE_M divides 1024

  // Build X = [obs(3) | latent(256) | zeros(29)] in bf16
  for (int i = tid; i < TILE_M * K1PAD; i += THREADS) {
    int m = i / K1PAD, k = i - m * K1PAD;
    float v;
    if (k < 3)        v = obs[(size_t)(p0 + m) * 3 + k];
    else if (k < 259) v = latent[batch * LATENT + (k - 3)];
    else              v = 0.0f;
    s_x[i] = (__bf16)v;
  }
  __syncthreads();
  layer_wmma<K1PAD, 512, true,  false>(s_x,  s_h1, W1t, b1, wave, lane);
  __syncthreads();
  layer_wmma<512,   512, true,  false>(s_h1, s_h2, W2t, b2, wave, lane);
  __syncthreads();
  layer_wmma<512,   256, true,  true >(s_h2, s_h3, W3t, b3, wave, lane);
  __syncthreads();
  // Layer 4: [256 -> 3], scalar f32 (N=3 too small for WMMA)
  if (tid < TILE_M * 3) {
    int p = tid & (TILE_M - 1);
    int d = tid / TILE_M;        // 0..2
    float s = b4[d];
    const float* hp = s_h3 + p * 256;
#pragma unroll 4
    for (int k = 0; k < 256; ++k) s = fmaf(hp[k], W4[k * 3 + d], s);
    size_t gi = (size_t)(p0 + p) * 3 + d;
    obs_est[gi] = obs[gi] + s;   // obs + delta
  }
}

// ---- chamfer + L2 loss, one block per batch; deterministic partials ----
__global__ __launch_bounds__(THREADS)
void loss_kernel(const float* __restrict__ obs_gt, const float* __restrict__ obs_est,
                 float* __restrict__ partials) {
  __shared__ float sx[NSAMP * 3];   // gt
  __shared__ float sy[NSAMP * 3];   // est
  __shared__ float red[THREADS];
  const int tid = threadIdx.x, b = blockIdx.x;
  const float* gx = obs_gt  + (size_t)b * NSAMP * 3;
  const float* gy = obs_est + (size_t)b * NSAMP * 3;
  for (int i = tid; i < NSAMP * 3; i += THREADS) { sx[i] = gx[i]; sy[i] = gy[i]; }
  __syncthreads();

  float chm = 0.0f, l2 = 0.0f;
  for (int n = tid; n < NSAMP; n += THREADS) {           // min over est (m)
    float x0 = sx[3*n], x1 = sx[3*n+1], x2 = sx[3*n+2];
    float best = 3.4e38f;
    for (int m = 0; m < NSAMP; ++m) {
      float d0 = x0 - sy[3*m], d1 = x1 - sy[3*m+1], d2 = x2 - sy[3*m+2];
      best = fminf(best, fmaf(d0, d0, fmaf(d1, d1, d2 * d2)));
    }
    chm += best;
    float e0 = x0 - sy[3*n], e1 = x1 - sy[3*n+1], e2 = x2 - sy[3*n+2];
    l2 += fmaf(e0, e0, fmaf(e1, e1, e2 * e2));
  }
  for (int m = tid; m < NSAMP; m += THREADS) {           // min over gt (n)
    float y0 = sy[3*m], y1 = sy[3*m+1], y2 = sy[3*m+2];
    float best = 3.4e38f;
    for (int n = 0; n < NSAMP; ++n) {
      float d0 = sx[3*n] - y0, d1 = sx[3*n+1] - y1, d2 = sx[3*n+2] - y2;
      best = fminf(best, fmaf(d0, d0, fmaf(d1, d1, d2 * d2)));
    }
    chm += best;
  }
  // 0.2 * (mean_n + mean_m) + 0.8 * mean(l2); denominators B*N=32768, B*N*3=98304
  red[tid] = 0.2f * chm * (1.0f / 32768.0f) + 0.8f * l2 * (1.0f / 98304.0f);
  __syncthreads();
  for (int s = THREADS / 2; s > 0; s >>= 1) {
    if (tid < s) red[tid] += red[tid + s];
    __syncthreads();
  }
  if (tid == 0) partials[b] = red[0];
}

__global__ void finalize_kernel(const float* __restrict__ partials, float* __restrict__ out) {
  if (threadIdx.x == 0 && blockIdx.x == 0) {
    float s = 0.0f;
    for (int i = 0; i < BATCHES; ++i) s += partials[i];  // fixed order -> deterministic
    out[0] = s;
  }
}

extern "C" void kernel_launch(void* const* d_in, const int* in_sizes, int n_in,
                              void* d_out, int out_size, void* d_ws, size_t ws_size,
                              hipStream_t stream) {
  (void)in_sizes; (void)n_in; (void)out_size; (void)ws_size;
  const float* obs    = (const float*)d_in[0];
  const float* obs_gt = (const float*)d_in[1];
  const float* latent = (const float*)d_in[2];
  const float* W1 = (const float*)d_in[3];  const float* b1 = (const float*)d_in[4];
  const float* W2 = (const float*)d_in[5];  const float* b2 = (const float*)d_in[6];
  const float* W3 = (const float*)d_in[7];  const float* b3 = (const float*)d_in[8];
  const float* W4 = (const float*)d_in[9];  const float* b4 = (const float*)d_in[10];

  char* ws = (char*)d_ws;
  float*  partials = (float*)ws;                          // 32 floats (pad to 256B)
  float*  obs_est  = (float*)(ws + 256);                  // 32768*3*4 = 393216 B
  __bf16* W1t = (__bf16*)(ws + 256 + 393216);             // 512*288*2 = 294912 B
  __bf16* W2t = (__bf16*)((char*)W1t + 512 * K1PAD * 2);  // 512*512*2 = 524288 B
  __bf16* W3t = (__bf16*)((char*)W2t + 512 * 512 * 2);    // 256*512*2 = 262144 B

  const int preptot = 512 * K1PAD + 512 * 512 + 256 * 512;
  prep_kernel<<<(preptot + THREADS - 1) / THREADS, THREADS, 0, stream>>>(
      W1, W2, W3, W1t, W2t, W3t);
  mlp_kernel<<<NPTS / TILE_M, THREADS, 0, stream>>>(
      obs, latent, W1t, b1, W2t, b2, W3t, b3, W4, b4, obs_est);
  loss_kernel<<<BATCHES, THREADS, 0, stream>>>(obs_gt, obs_est, partials);
  finalize_kernel<<<1, 32, 0, stream>>>(partials, (float*)d_out);
}
